// SequentialMLP_46007689675055
// MI455X (gfx1250) — matrix operand order repaired and checked by reading, not compiled, not run on hardware
//
#include <hip/hip_runtime.h>
#include <hip/hip_bf16.h>
#include <cstdint>
#include <cstddef>

typedef __attribute__((ext_vector_type(16))) __bf16 v16bf;
typedef __attribute__((ext_vector_type(8)))  __bf16 v8bf;
typedef __attribute__((ext_vector_type(4)))  __bf16 v4bf;
typedef __attribute__((ext_vector_type(2)))  __bf16 v2bf;
typedef __attribute__((ext_vector_type(8)))  float  v8f;

#define NEXP  8
#define TOK   8192
#define HID   2048
#define INTER 5632

#define BM 128
#define BN 64
#define BK 32
#define AS 40   // A LDS row stride in bf16 elems (32 data + 8 pad -> 80B, 16B-aligned, bank-friendly)
#define BS 40   // B (transposed) LDS row stride
#define ASZ (BM * AS)   // one A buffer
#define BSZ (BN * BS)   // one B buffer

static_assert((HID / BK) % 2 == 0, "k-loop unroll-by-2 needs even NK");
static_assert((INTER / BK) % 2 == 0, "k-loop unroll-by-2 needs even NK");

// Split workgroup barrier (CDNA5): drain only DScnt (LDS stores/loads) before
// rendezvous, leaving global loads in flight across the barrier. __syncthreads()
// would conservatively wait loadcnt==0 too, killing the k+2 prefetch distance.
__device__ __forceinline__ void wg_barrier_lds() {
    asm volatile("s_wait_dscnt 0x0\n\t"
                 "s_barrier_signal -1\n\t"
                 "s_barrier_wait -1" ::: "memory");
}

// Tokens are pre-sorted by expert; find expert owning row0 from device-side counts.
__device__ __forceinline__ int find_expert(const int* __restrict__ tpe, int row0) {
    int acc = 0, e = 0;
#pragma unroll
    for (int i = 0; i < NEXP; ++i) {
        acc += tpe[i];
        e += (row0 >= acc) ? 1 : 0;
    }
    return (e < NEXP) ? e : (NEXP - 1);
}

// A-matrix 16x32 bf16 fragment (ISA 7.12.2): lane = 16*khalf + m.
__device__ __forceinline__ v16bf ld_a_frag(const __bf16* A, int row, int half) {
    const v8bf lo = *(const v8bf*)(A + row * AS + half * 8);
    const v8bf hi = *(const v8bf*)(A + row * AS + 16 + half * 8);
    v16bf r;
#pragma unroll
    for (int i = 0; i < 8; ++i) { r[i] = lo[i]; r[i + 8] = hi[i]; }
    return r;
}

// B-matrix 32x16 bf16 fragment: lanes 0-15 hold K=0..7,16..23 of column N=lane%16,
// lanes 16-31 hold K=8..15,24..31. LDS holds B transposed [n][k].
__device__ __forceinline__ v16bf ld_b_frag(const __bf16* B, int n, int khalf) {
    const v8bf lo = *(const v8bf*)(B + n * BS + khalf * 8);
    const v8bf hi = *(const v8bf*)(B + n * BS + 16 + khalf * 8);
    v16bf r;
#pragma unroll
    for (int i = 0; i < 8; ++i) { r[i] = lo[i]; r[i + 8] = hi[i]; }
    return r;
}

// ---- staging helpers: global->regs (two k-iters ahead) and regs->LDS ----
__device__ __forceinline__ void a_glb(const float* __restrict__ x, int row0, int k0,
                                      int tid, float4 f[4]) {
#pragma unroll
    for (int j = 0; j < 4; ++j) {
        const int idx = tid + j * 256;
        const int m = idx >> 3, c = idx & 7;
        f[j] = *(const float4*)(x + (size_t)(row0 + m) * HID + k0 + c * 4);
    }
}
__device__ __forceinline__ void a_lds(__bf16* Alds, int tid, const float4 f[4]) {
#pragma unroll
    for (int j = 0; j < 4; ++j) {
        const int idx = tid + j * 256;
        const int m = idx >> 3, c = idx & 7;
        v4bf p = { (__bf16)f[j].x, (__bf16)f[j].y, (__bf16)f[j].z, (__bf16)f[j].w };
        *(v4bf*)(Alds + m * AS + c * 4) = p;
    }
}
__device__ __forceinline__ void b_glb(const float* __restrict__ w, int ld, int col0,
                                      int k0, int sn, int kb, float f0[4], float f1[4]) {
#pragma unroll
    for (int p = 0; p < 4; ++p) {
        const int k = kb + p * 8;
        const size_t gi = (size_t)(k0 + k) * ld + col0 + sn;
        f0[p] = w[gi];
        f1[p] = w[gi + ld];
    }
}
__device__ __forceinline__ void b_lds(__bf16* B, int sn, int kb,
                                      const float f0[4], const float f1[4]) {
#pragma unroll
    for (int p = 0; p < 4; ++p) {
        const int k = kb + p * 8;
        v2bf pk = { (__bf16)f0[p], (__bf16)f1[p] };
        *(v2bf*)(B + sn * BS + k) = pk;
    }
}

// -------- Kernel 1: h = silu(x*Wg[e]) * (x*Wu[e]), h stored bf16 [T][I] --------
__global__ __launch_bounds__(256, 2)
void moe_gateup(const float* __restrict__ x, const int* __restrict__ tpe,
                const float* __restrict__ wg, const float* __restrict__ wu,
                __bf16* __restrict__ hb)
{
    __shared__ __align__(16) __bf16 Alds[2 * ASZ];
    __shared__ __align__(16) __bf16 Bgl[2 * BSZ];
    __shared__ __align__(16) __bf16 Bul[2 * BSZ];

    const int tid  = threadIdx.x;
    const int lane = tid & 31;
    const int wave = tid >> 5;
    const int wm   = (wave >> 1) * 32;
    const int wn   = (wave & 1) * 32;
    const int half = lane >> 4;
    const int l16  = lane & 15;

    const int row0 = blockIdx.y * BM;
    const int col0 = blockIdx.x * BN;

    const int e = find_expert(tpe, row0);
    const float* wgp = wg + (size_t)e * HID * INTER;
    const float* wup = wu + (size_t)e * HID * INTER;

    v8f accg[2][2], accu[2][2];
#pragma unroll
    for (int i = 0; i < 2; ++i)
#pragma unroll
        for (int j = 0; j < 2; ++j)
#pragma unroll
            for (int v = 0; v < 8; ++v) { accg[i][j][v] = 0.0f; accu[i][j][v] = 0.0f; }

    const int sn = tid & 63;
    const int kb = (tid >> 6) * 2;

    constexpr int NK = HID / BK;

    // pipeline prologue: tile 0 -> regs -> LDS buf0; tile 1 -> regs
    float4 af[4];
    float bg0[4], bg1[4], bu0[4], bu1[4];
    a_glb(x, row0, 0, tid, af);
    b_glb(wgp, INTER, col0, 0, sn, kb, bg0, bg1);
    b_glb(wup, INTER, col0, 0, sn, kb, bu0, bu1);
    a_lds(Alds, tid, af);
    b_lds(Bgl, sn, kb, bg0, bg1);
    b_lds(Bul, sn, kb, bu0, bu1);
    a_glb(x, row0, BK, tid, af);
    b_glb(wgp, INTER, col0, BK, sn, kb, bg0, bg1);
    b_glb(wup, INTER, col0, BK, sn, kb, bu0, bu1);
    wg_barrier_lds();

    // one pipeline step; buffer bases are compile-time-constant per call site
    auto step = [&](const __bf16* Ac, const __bf16* Gc, const __bf16* Uc,
                    __bf16* An, __bf16* Gn, __bf16* Un, int kk) {
        const int k0 = kk * BK;

        // frag loads from current buffer (DS in-order: these complete first)
        const v16bf a0 = ld_a_frag(Ac, wm + l16, half);
        const v16bf a1 = ld_a_frag(Ac, wm + 16 + l16, half);
        const v16bf g0 = ld_b_frag(Gc, wn + l16, half);
        const v16bf g1 = ld_b_frag(Gc, wn + 16 + l16, half);
        const v16bf u0 = ld_b_frag(Uc, wn + l16, half);
        const v16bf u1 = ld_b_frag(Uc, wn + 16 + l16, half);

        // drain regs (tile kk+1) into the other buffer; overlaps with WMMAs
        if (kk + 1 < NK) {
            a_lds(An, tid, af);
            b_lds(Gn, sn, kb, bg0, bg1);
            b_lds(Un, sn, kb, bu0, bu1);
        }
        // issue tile kk+2 global loads; two full iterations to land
        if (kk + 2 < NK) {
            a_glb(x, row0, k0 + 2 * BK, tid, af);
            b_glb(wgp, INTER, col0, k0 + 2 * BK, sn, kb, bg0, bg1);
            b_glb(wup, INTER, col0, k0 + 2 * BK, sn, kb, bu0, bu1);
        }
        if (kk + 3 < NK) {  // warm L2 three tiles ahead
            __builtin_prefetch(wgp + (size_t)(k0 + 3 * BK + kb) * INTER + col0 + sn, 0, 1);
            __builtin_prefetch(wup + (size_t)(k0 + 3 * BK + kb) * INTER + col0 + sn, 0, 1);
        }

        accg[0][0] = __builtin_amdgcn_wmma_f32_16x16x32_bf16(false, a0, false, g0, (short)0, accg[0][0], false, false);
        accg[0][1] = __builtin_amdgcn_wmma_f32_16x16x32_bf16(false, a0, false, g1, (short)0, accg[0][1], false, false);
        accg[1][0] = __builtin_amdgcn_wmma_f32_16x16x32_bf16(false, a1, false, g0, (short)0, accg[1][0], false, false);
        accg[1][1] = __builtin_amdgcn_wmma_f32_16x16x32_bf16(false, a1, false, g1, (short)0, accg[1][1], false, false);
        accu[0][0] = __builtin_amdgcn_wmma_f32_16x16x32_bf16(false, a0, false, u0, (short)0, accu[0][0], false, false);
        accu[0][1] = __builtin_amdgcn_wmma_f32_16x16x32_bf16(false, a0, false, u1, (short)0, accu[0][1], false, false);
        accu[1][0] = __builtin_amdgcn_wmma_f32_16x16x32_bf16(false, a1, false, u0, (short)0, accu[1][0], false, false);
        accu[1][1] = __builtin_amdgcn_wmma_f32_16x16x32_bf16(false, a1, false, u1, (short)0, accu[1][1], false, false);

        wg_barrier_lds();   // nxt stores visible; cur free for tile kk+2
    };

    for (int kk = 0; kk < NK; kk += 2) {
        step(Alds,       Bgl,       Bul,       Alds + ASZ, Bgl + BSZ, Bul + BSZ, kk);
        step(Alds + ASZ, Bgl + BSZ, Bul + BSZ, Alds,       Bgl,       Bul,       kk + 1);
    }

    // Epilogue: h = silu(g) * u, bf16. C/D layout: M = v + 8*half, N = l16.
#pragma unroll
    for (int ms = 0; ms < 2; ++ms)
#pragma unroll
        for (int ns = 0; ns < 2; ++ns)
#pragma unroll
            for (int v = 0; v < 8; ++v) {
                const float g = accg[ms][ns][v];
                const float u = accu[ms][ns][v];
                const float s = g / (1.0f + __expf(-g));
                const int row = row0 + wm + ms * 16 + v + 8 * half;
                const int col = col0 + wn + ns * 16 + l16;
                hb[(size_t)row * INTER + col] = (__bf16)(s * u);
            }
}

// -------- Kernel 2: out = h * Wd[e], f32 [T][H] --------
__global__ __launch_bounds__(256, 2)
void moe_down(const __bf16* __restrict__ hb, const int* __restrict__ tpe,
              const float* __restrict__ wd, float* __restrict__ out)
{
    __shared__ __align__(16) __bf16 Alds[2 * ASZ];
    __shared__ __align__(16) __bf16 Bdl[2 * BSZ];

    const int tid  = threadIdx.x;
    const int lane = tid & 31;
    const int wave = tid >> 5;
    const int wm   = (wave >> 1) * 32;
    const int wn   = (wave & 1) * 32;
    const int half = lane >> 4;
    const int l16  = lane & 15;

    const int row0 = blockIdx.y * BM;
    const int col0 = blockIdx.x * BN;

    const int e = find_expert(tpe, row0);
    const float* wdp = wd + (size_t)e * INTER * HID;

    v8f acc[2][2];
#pragma unroll
    for (int i = 0; i < 2; ++i)
#pragma unroll
        for (int j = 0; j < 2; ++j)
#pragma unroll
            for (int v = 0; v < 8; ++v) acc[i][j][v] = 0.0f;

    const int sn = tid & 63;
    const int kb = (tid >> 6) * 2;

    constexpr int NK = INTER / BK;

    // pipeline prologue
    v8bf ah[2];
    float bd0[4], bd1[4];
#pragma unroll
    for (int j = 0; j < 2; ++j) {
        const int idx = tid + j * 256;
        const int m = idx >> 2, c = idx & 3;
        ah[j] = *(const v8bf*)(hb + (size_t)(row0 + m) * INTER + c * 8);
    }
    b_glb(wdp, HID, col0, 0, sn, kb, bd0, bd1);
#pragma unroll
    for (int j = 0; j < 2; ++j) {
        const int idx = tid + j * 256;
        const int m = idx >> 2, c = idx & 3;
        *(v8bf*)(Alds + m * AS + c * 8) = ah[j];
    }
    b_lds(Bdl, sn, kb, bd0, bd1);
#pragma unroll
    for (int j = 0; j < 2; ++j) {
        const int idx = tid + j * 256;
        const int m = idx >> 2, c = idx & 3;
        ah[j] = *(const v8bf*)(hb + (size_t)(row0 + m) * INTER + BK + c * 8);
    }
    b_glb(wdp, HID, col0, BK, sn, kb, bd0, bd1);
    wg_barrier_lds();

    auto step = [&](const __bf16* Ac, const __bf16* Dc,
                    __bf16* An, __bf16* Dn, int kk) {
        const int k0 = kk * BK;

        const v16bf a0 = ld_a_frag(Ac, wm + l16, half);
        const v16bf a1 = ld_a_frag(Ac, wm + 16 + l16, half);
        const v16bf d0 = ld_b_frag(Dc, wn + l16, half);
        const v16bf d1 = ld_b_frag(Dc, wn + 16 + l16, half);

        if (kk + 1 < NK) {
#pragma unroll
            for (int j = 0; j < 2; ++j) {
                const int idx = tid + j * 256;
                const int m = idx >> 2, c = idx & 3;
                *(v8bf*)(An + m * AS + c * 8) = ah[j];
            }
            b_lds(Dn, sn, kb, bd0, bd1);
        }
        if (kk + 2 < NK) {
#pragma unroll
            for (int j = 0; j < 2; ++j) {
                const int idx = tid + j * 256;
                const int m = idx >> 2, c = idx & 3;
                ah[j] = *(const v8bf*)(hb + (size_t)(row0 + m) * INTER + k0 + 2 * BK + c * 8);
            }
            b_glb(wdp, HID, col0, k0 + 2 * BK, sn, kb, bd0, bd1);
        }
        if (kk + 3 < NK) {
            __builtin_prefetch(wdp + (size_t)(k0 + 3 * BK + kb) * HID + col0 + sn, 0, 1);
        }

        acc[0][0] = __builtin_amdgcn_wmma_f32_16x16x32_bf16(false, a0, false, d0, (short)0, acc[0][0], false, false);
        acc[0][1] = __builtin_amdgcn_wmma_f32_16x16x32_bf16(false, a0, false, d1, (short)0, acc[0][1], false, false);
        acc[1][0] = __builtin_amdgcn_wmma_f32_16x16x32_bf16(false, a1, false, d0, (short)0, acc[1][0], false, false);
        acc[1][1] = __builtin_amdgcn_wmma_f32_16x16x32_bf16(false, a1, false, d1, (short)0, acc[1][1], false, false);

        wg_barrier_lds();
    };

    for (int kk = 0; kk < NK; kk += 2) {
        step(Alds,       Bdl,       Alds + ASZ, Bdl + BSZ, kk);
        step(Alds + ASZ, Bdl + BSZ, Alds,       Bdl,       kk + 1);
    }

#pragma unroll
    for (int ms = 0; ms < 2; ++ms)
#pragma unroll
        for (int ns = 0; ns < 2; ++ns)
#pragma unroll
            for (int v = 0; v < 8; ++v) {
                const int row = row0 + wm + ms * 16 + v + 8 * half;
                const int col = col0 + wn + ns * 16 + l16;
                out[(size_t)row * HID + col] = acc[ms][ns][v];
            }
}

extern "C" void kernel_launch(void* const* d_in, const int* in_sizes, int n_in,
                              void* d_out, int out_size, void* d_ws, size_t ws_size,
                              hipStream_t stream) {
    const float* x   = (const float*)d_in[0];
    const int*   tpe = (const int*)d_in[1];
    const float* wg  = (const float*)d_in[2];
    const float* wu  = (const float*)d_in[3];
    const float* wd  = (const float*)d_in[4];
    float* out = (float*)d_out;
    __bf16* hb = (__bf16*)d_ws;   // bf16 intermediate h: TOK x INTER (~92 MB)

    dim3 blk(256);
    dim3 g1(INTER / BN, TOK / BM);   // 88 x 64 blocks
    moe_gateup<<<g1, blk, 0, stream>>>(x, tpe, wg, wu, hb);

    dim3 g2(HID / BN, TOK / BM);     // 32 x 64 blocks
    moe_down<<<g2, blk, 0, stream>>>(hb, tpe, wd, out);
}
